// GCT_28389733826937
// MI455X (gfx1250) — matrix operand order repaired and checked by reading, not compiled, not run on hardware
//
#include <hip/hip_runtime.h>
#include <hip/hip_bf16.h>

typedef __attribute__((ext_vector_type(16))) _Float16 v16h;
typedef __attribute__((ext_vector_type(8)))  _Float16 v8h;
typedef __attribute__((ext_vector_type(8)))  float    v8f;

// ---------------------------------------------------------------------------
// Generic GEMM: C[M,N] = act( op(A)[M,K] * op(B)[K,N] + bias )
// A element (m,k) = A[m*as0 + k*as1]; B element (k,n) = B[k*bs0 + n*bs1]
// biasMode: 0=none, 1=bias[m], 2=bias[n]; act: 0=none, 1=leaky_relu(0.01)
// Block = 128 threads (4 waves). Tile: 64(M) x 64(N), K-step 32, f16 WMMA.
// Interior blocks + full K-steps take a predicate-free staging path chosen by
// a block-uniform scalar branch (no exec-mask churn).
// ---------------------------------------------------------------------------
__global__ __launch_bounds__(128)
void k_gemm_wmma(const float* __restrict__ A, int as0, int as1,
                 const float* __restrict__ B, int bs0, int bs1,
                 float* __restrict__ C, const float* __restrict__ bias,
                 int biasMode, int M, int N, int K, int act) {
  __shared__ __align__(16) _Float16 As[64 * 32];
  __shared__ __align__(16) _Float16 Bt[64 * 32];   // stored [n][k], K-contiguous

  const int tid  = threadIdx.x;
  const int wave = tid >> 5;
  const int lane = tid & 31;
  const int rowBase = blockIdx.y * 64;
  const int colBase = blockIdx.x * 64;
  const bool tileFull = (rowBase + 64 <= M) && (colBase + 64 <= N);

  // staging coordinates (fixed per thread)
  const int r    = tid >> 1;           // 0..63
  const int kk0  = (tid & 1) * 16;     // 0 or 16
  const int mg   = rowBase + r;
  const int ng   = colBase + r;
  const bool mOK = (mg < M);
  const bool nOK = (ng < N);
  const int mgc  = mOK ? mg : 0;
  const int ngc  = nOK ? ng : 0;
  // vector-path legality for A: K-contiguous rows, 16B-aligned row starts
  const bool aVec = (as1 == 1) && ((as0 & 3) == 0);

  v8f acc[4];
  #pragma unroll
  for (int j = 0; j < 4; ++j) acc[j] = (v8f){0.f,0.f,0.f,0.f,0.f,0.f,0.f,0.f};

  const int ksteps = (K + 31) / 32;
  for (int ks = 0; ks < ksteps; ++ks) {
    const int k0  = ks * 32;
    const int kg0 = k0 + kk0;

    float av[16], bv[16];
    if (tileFull && (k0 + 32 <= K)) {
      // ---------- predicate-free fast path (block-uniform branch) ----------
      if (aVec) {
        const float4* p = (const float4*)(A + (size_t)(mg * as0 + kg0));
        #pragma unroll
        for (int q = 0; q < 4; ++q) {
          float4 v = p[q];
          av[4*q+0] = v.x; av[4*q+1] = v.y; av[4*q+2] = v.z; av[4*q+3] = v.w;
        }
      } else {
        const float* ap = A + (size_t)(mg * as0 + kg0 * as1);
        #pragma unroll
        for (int i = 0; i < 16; ++i) av[i] = ap[(size_t)i * as1];
      }
      const float* bp = B + (size_t)(kg0 * bs0 + ng * bs1);
      #pragma unroll
      for (int i = 0; i < 16; ++i) bv[i] = bp[(size_t)i * bs0];
    } else {
      // ---------- guarded path (boundary tiles / K tail) ----------
      #pragma unroll
      for (int i = 0; i < 16; ++i) {
        const int kg = kg0 + i;
        const bool ok = mOK & (kg < K);
        float t = A[(size_t)(mgc * as0 + (ok ? kg : 0) * as1)];
        av[i] = ok ? t : 0.f;
      }
      #pragma unroll
      for (int i = 0; i < 16; ++i) {
        const int kg = kg0 + i;
        const bool ok = nOK & (kg < K);
        float t = B[(size_t)((ok ? kg : 0) * bs0 + ngc * bs1)];
        bv[i] = ok ? t : 0.f;
      }
    }

    // convert + vectorized LDS stores (2 x b128 each)
    v8h a0, a1, b0, b1;
    #pragma unroll
    for (int i = 0; i < 8; ++i) {
      a0[i] = (_Float16)av[i];     a1[i] = (_Float16)av[8 + i];
      b0[i] = (_Float16)bv[i];     b1[i] = (_Float16)bv[8 + i];
    }
    *(v8h*)&As[r * 32 + kk0]     = a0;
    *(v8h*)&As[r * 32 + kk0 + 8] = a1;
    *(v8h*)&Bt[r * 32 + kk0]     = b0;
    *(v8h*)&Bt[r * 32 + kk0 + 8] = b1;

    if (ks + 1 < ksteps && mOK)   // global_prefetch_b8 hint for next A K-tile
      __builtin_prefetch((const void*)(A + (size_t)(mgc * as0 + (k0 + 32) * as1)), 0, 1);
    __syncthreads();

    // ---- A fragment (16x32, k = 16*(lane/16) + s, same order as B): two aligned 16B LDS loads ----
    const int mL   = wave * 16 + (lane & 15);
    const int aoff = (lane >= 16) ? 16 : 0;
    const v8h a_lo = *(const v8h*)&As[mL * 32 + aoff];
    const v8h a_hi = *(const v8h*)&As[mL * 32 + 8 + aoff];
    v16h a;
    #pragma unroll
    for (int i = 0; i < 8; ++i) { a[i] = a_lo[i]; a[8 + i] = a_hi[i]; }

    // ---- 4 WMMAs over the 64-wide N tile ----
    #pragma unroll
    for (int j = 0; j < 4; ++j) {
      const int nL = j * 16 + (lane & 15);
      const v16h b = *(const v16h*)&Bt[nL * 32 + ((lane >= 16) ? 16 : 0)];
      acc[j] = __builtin_amdgcn_wmma_f32_16x16x32_f16(
          false, a, false, b, (short)0, acc[j], false, false);
    }
    __syncthreads();
  }

  // ---- epilogue: C/D layout -> global, fused bias + activation ----
  const int nLane = lane & 15;
  const int mAdd  = (lane >= 16) ? 8 : 0;
  #pragma unroll
  for (int j = 0; j < 4; ++j) {
    const int ngo = colBase + j * 16 + nLane;
    if (!tileFull && ngo >= N) continue;
    #pragma unroll
    for (int rr = 0; rr < 8; ++rr) {
      const int mgo = rowBase + wave * 16 + rr + mAdd;
      if (!tileFull && mgo >= M) continue;
      float v = acc[j][rr];
      if (biasMode == 1) v += bias[mgo];
      else if (biasMode == 2) v += bias[ngo];
      if (act == 1) v = v > 0.f ? v : 0.01f * v;
      C[(size_t)mgo * N + ngo] = v;
    }
  }
}

// ---------------------------------------------------------------------------
// Implicit-GEMM conv2d: Y[n,oc,oh,ow] = relu( W[oc,:]*im2col + bias[oc] )
// GEMM view: M=OC, N=Nimg*OH*OW, K=C*KH*KW. Pixel decomposition hoisted,
// (c,kh,kw) tracked incrementally; interior tiles use predicate-free weights
// staging and pad-only checks on the im2col gather.
// ---------------------------------------------------------------------------
__global__ __launch_bounds__(128)
void k_conv2d_wmma(const float* __restrict__ X, const float* __restrict__ Wt,
                   const float* __restrict__ bias, float* __restrict__ Y,
                   int Nimg, int Cin, int H, int Wd, int OC, int KH, int KW,
                   int stride, int pad, int OH, int OW, int relu) {
  __shared__ __align__(16) _Float16 As[64 * 32];
  __shared__ __align__(16) _Float16 Bt[64 * 32];

  const int tid  = threadIdx.x;
  const int wave = tid >> 5;
  const int lane = tid & 31;
  const int rowBase = blockIdx.y * 64;           // OC
  const int colBase = blockIdx.x * 64;           // pixel columns
  const int K    = Cin * KH * KW;
  const int cols = Nimg * OH * OW;
  const int khw  = KH * KW;
  const int ohw  = OH * OW;
  const bool tileFull = (rowBase + 64 <= OC) && (colBase + 64 <= cols);

  // staging coordinates (fixed per thread) + hoisted pixel decomposition
  const int r    = tid >> 1;
  const int kk0  = (tid & 1) * 16;
  const int mg   = rowBase + r;                  // oc
  const int ngB  = colBase + r;                  // pixel col
  const bool mOK = (mg < OC);
  const bool nOK = (ngB < cols);
  const int ngc  = nOK ? ngB : 0;
  const int nImg = ngc / ohw;
  const int pPix = ngc - nImg * ohw;
  const int ohP  = pPix / OW;
  const int owP  = pPix - ohP * OW;
  const int ihBase = ohP * stride - pad;
  const int iwBase = owP * stride - pad;
  const float* Xn = X + (size_t)nImg * Cin * H * Wd;
  const bool aVec = ((K & 3) == 0);

  v8f acc[4];
  #pragma unroll
  for (int j = 0; j < 4; ++j) acc[j] = (v8f){0.f,0.f,0.f,0.f,0.f,0.f,0.f,0.f};

  const int ksteps = (K + 31) / 32;
  for (int ks = 0; ks < ksteps; ++ks) {
    const int k0  = ks * 32;
    const int kg0 = k0 + kk0;

    float av[16], bv[16];
    if (tileFull && (k0 + 32 <= K)) {
      // ---------- fast path: no M/N/K guards (block-uniform) ----------
      if (aVec) {
        const float4* p = (const float4*)(Wt + (size_t)(mg * K + kg0));
        #pragma unroll
        for (int q = 0; q < 4; ++q) {
          float4 v = p[q];
          av[4*q+0] = v.x; av[4*q+1] = v.y; av[4*q+2] = v.z; av[4*q+3] = v.w;
        }
      } else {
        const float* wp = Wt + (size_t)(mg * K + kg0);
        #pragma unroll
        for (int i = 0; i < 16; ++i) av[i] = wp[i];
      }
      int c   = kg0 / khw;
      int rem = kg0 - c * khw;
      int kh  = rem / KW;
      int kw  = rem - kh * KW;
      #pragma unroll
      for (int i = 0; i < 16; ++i) {
        const int ih = ihBase + kh;
        const int iw = iwBase + kw;
        const bool inb = (ih >= 0) & (ih < H) & (iw >= 0) & (iw < Wd);
        const int idx = inb ? ((c * H + ih) * Wd + iw) : 0;
        float t = Xn[idx];
        bv[i] = inb ? t : 0.f;
        ++kw;
        if (kw == KW) { kw = 0; ++kh; if (kh == KH) { kh = 0; ++c; } }
      }
    } else {
      // ---------- guarded path ----------
      #pragma unroll
      for (int i = 0; i < 16; ++i) {
        const int kg = kg0 + i;
        const bool ok = mOK & (kg < K);
        float t = Wt[(size_t)((mOK ? mg : 0) * K + (ok ? kg : 0))];
        av[i] = ok ? t : 0.f;
      }
      int c   = kg0 / khw;
      int rem = kg0 - c * khw;
      int kh  = rem / KW;
      int kw  = rem - kh * KW;
      #pragma unroll
      for (int i = 0; i < 16; ++i) {
        const int kg = kg0 + i;
        const int ih = ihBase + kh;
        const int iw = iwBase + kw;
        const bool inb = nOK & (kg < K) & (ih >= 0) & (ih < H) & (iw >= 0) & (iw < Wd);
        const int idx = inb ? ((c * H + ih) * Wd + iw) : 0;
        float t = Xn[idx];
        bv[i] = inb ? t : 0.f;
        ++kw;
        if (kw == KW) { kw = 0; ++kh; if (kh == KH) { kh = 0; ++c; } }
      }
    }

    v8h a0, a1, b0, b1;
    #pragma unroll
    for (int i = 0; i < 8; ++i) {
      a0[i] = (_Float16)av[i];     a1[i] = (_Float16)av[8 + i];
      b0[i] = (_Float16)bv[i];     b1[i] = (_Float16)bv[8 + i];
    }
    *(v8h*)&As[r * 32 + kk0]     = a0;
    *(v8h*)&As[r * 32 + kk0 + 8] = a1;
    *(v8h*)&Bt[r * 32 + kk0]     = b0;
    *(v8h*)&Bt[r * 32 + kk0 + 8] = b1;

    if (ks + 1 < ksteps && mOK)
      __builtin_prefetch((const void*)(Wt + (size_t)(mg * K + k0 + 32)), 0, 1);
    __syncthreads();

    const int mL   = wave * 16 + (lane & 15);
    const int aoff = (lane >= 16) ? 16 : 0;
    const v8h a_lo = *(const v8h*)&As[mL * 32 + aoff];
    const v8h a_hi = *(const v8h*)&As[mL * 32 + 8 + aoff];
    v16h a;
    #pragma unroll
    for (int i = 0; i < 8; ++i) { a[i] = a_lo[i]; a[8 + i] = a_hi[i]; }

    #pragma unroll
    for (int j = 0; j < 4; ++j) {
      const int nL = j * 16 + (lane & 15);
      const v16h b = *(const v16h*)&Bt[nL * 32 + ((lane >= 16) ? 16 : 0)];
      acc[j] = __builtin_amdgcn_wmma_f32_16x16x32_f16(
          false, a, false, b, (short)0, acc[j], false, false);
    }
    __syncthreads();
  }

  const int nLane = lane & 15;
  const int mAdd  = (lane >= 16) ? 8 : 0;
  #pragma unroll
  for (int j = 0; j < 4; ++j) {
    const int ngo = colBase + j * 16 + nLane;
    if (!tileFull && ngo >= cols) continue;
    const int n = ngo / ohw;
    const int p = ngo - n * ohw;
    #pragma unroll
    for (int rr = 0; rr < 8; ++rr) {
      const int oc = rowBase + wave * 16 + rr + mAdd;
      if (!tileFull && oc >= OC) continue;
      float v = acc[j][rr] + bias[oc];
      if (relu) v = fmaxf(v, 0.f);
      Y[((size_t)n * OC + oc) * ohw + p] = v;
    }
  }
}

// ---------------------------------------------------------------------------
// 3x3 stride-2 VALID max pool
// ---------------------------------------------------------------------------
__global__ void k_maxpool3x3s2(const float* __restrict__ X, float* __restrict__ Y,
                               int NC, int H, int W, int OH, int OW) {
  int idx = blockIdx.x * blockDim.x + threadIdx.x;
  int tot = NC * OH * OW;
  if (idx >= tot) return;
  int ow = idx % OW; int t = idx / OW; int oh = t % OH; int nc = t / OH;
  const float* base = X + (size_t)nc * H * W + (oh * 2) * W + (ow * 2);
  float m = -3.4e38f;
  #pragma unroll
  for (int dh = 0; dh < 3; ++dh)
    #pragma unroll
    for (int dw = 0; dw < 3; ++dw)
      m = fmaxf(m, base[dh * W + dw]);
  Y[idx] = m;
}

// nodes[n=m*16+t][d] = Zf[t][d][m]  (Zf = FEAT images 1..16, each [256,88])
__global__ void k_build_nodes(const float* __restrict__ FEAT, float* __restrict__ NODES) {
  int idx = blockIdx.x * blockDim.x + threadIdx.x;
  if (idx >= 1408 * 256) return;
  int d = idx % 256; int n = idx / 256;
  int t = n & 15; int m = n >> 4;
  NODES[idx] = FEAT[(size_t)(1 + t) * 22528 + d * 88 + m];
}

// S[t][c] = sum_m P[(m*16+t)][c]
__global__ void k_colsum_t(const float* __restrict__ P, float* __restrict__ S, int CW) {
  int idx = blockIdx.x * blockDim.x + threadIdx.x;
  if (idx >= 16 * CW) return;
  int c = idx % CW; int t = idx / CW;
  float s = 0.f;
  for (int m = 0; m < 88; ++m) s += P[(size_t)(m * 16 + t) * CW + c];
  S[t * CW + c] = s;
}

// h[r][c] = lrelu( S[t][c] - (t==0)*P[r][c] + bias[c] ),  t = r%16
// (exploits A1 structure: same-t, diff-s adjacency + self-loops for t!=0)
__global__ void k_a1_apply(const float* __restrict__ P, const float* __restrict__ S,
                           const float* __restrict__ bias, float* __restrict__ Hout, int CW) {
  int idx = blockIdx.x * blockDim.x + threadIdx.x;
  if (idx >= 1408 * CW) return;
  int c = idx % CW; int r = idx / CW; int t = r & 15;
  float v = S[t * CW + c] - (t == 0 ? P[idx] : 0.f) + bias[c];
  Hout[idx] = v > 0.f ? v : 0.01f * v;
}

// V1[d][m] = max_t H2[(m*16+t)][d]
__global__ void k_temporal_max(const float* __restrict__ H2, float* __restrict__ V1) {
  int idx = blockIdx.x * blockDim.x + threadIdx.x;
  if (idx >= 22528) return;
  int m = idx % 88; int d = idx / 88;
  float mx = -3.4e38f;
  for (int t = 0; t < 16; ++t)
    mx = fmaxf(mx, H2[(size_t)(m * 16 + t) * 256 + d]);
  V1[d * 88 + m] = mx;
}

// v[o] = max_m ( conv1d(Xf, wcd, k=3, p=1)[o][m] + bcd[o] )
__global__ __launch_bounds__(128)
void k_conv1d_vmax(const float* __restrict__ Xf, const float* __restrict__ wcd,
                   const float* __restrict__ bcd, float* __restrict__ v) {
  int o = blockIdx.x; int m = threadIdx.x;
  float acc = -3.4e38f;
  if (m < 88) {
    float s = bcd[o];
    for (int i = 0; i < 256; ++i) {
      const float* wp = &wcd[(size_t)(o * 256 + i) * 3];
      const float* xp = &Xf[i * 88];
      if (m > 0)  s += xp[m - 1] * wp[0];
      s += xp[m] * wp[1];
      if (m < 87) s += xp[m + 1] * wp[2];
    }
    acc = s;
  }
  __shared__ float red[128];
  red[threadIdx.x] = acc; __syncthreads();
  for (int off = 64; off > 0; off >>= 1) {
    if (threadIdx.x < off) red[threadIdx.x] = fmaxf(red[threadIdx.x], red[threadIdx.x + off]);
    __syncthreads();
  }
  if (threadIdx.x == 0) v[o] = red[0];
}

// X_hat[o][k] = sum_i v[i] * wt[i][o][k] + bt[o]
__global__ void k_deconv(const float* __restrict__ vv, const float* __restrict__ wt,
                         const float* __restrict__ bt, float* __restrict__ XH) {
  int idx = blockIdx.x * blockDim.x + threadIdx.x;
  if (idx >= 22528) return;
  int k = idx % 88; int o = idx / 88;
  float s = bt[o];
  for (int i = 0; i < 256; ++i)
    s += vv[i] * wt[(size_t)(i * 256 + o) * 88 + k];
  XH[idx] = s;
}

__global__ void k_add(const float* __restrict__ a, const float* __restrict__ b,
                      float* __restrict__ c, int n) {
  int idx = blockIdx.x * blockDim.x + threadIdx.x;
  if (idx < n) c[idx] = a[idx] + b[idx];
}

// in-place row softmax on [88,88]
__global__ __launch_bounds__(128)
void k_softmax88(float* __restrict__ Sm) {
  int j = blockIdx.x; int i = threadIdx.x;
  __shared__ float red[128];
  float v = (i < 88) ? Sm[j * 88 + i] : -3.4e38f;
  red[i] = v; __syncthreads();
  for (int off = 64; off > 0; off >>= 1) {
    if (i < off) red[i] = fmaxf(red[i], red[i + off]);
    __syncthreads();
  }
  float mx = red[0]; __syncthreads();
  float e = (i < 88) ? expf(v - mx) : 0.f;
  red[i] = e; __syncthreads();
  for (int off = 64; off > 0; off >>= 1) {
    if (i < off) red[i] += red[i + off];
    __syncthreads();
  }
  if (i < 88) Sm[j * 88 + i] = e / red[0];
}

// R = sum_{d,m} Xf[d][m] * V2t[m][d]
__global__ __launch_bounds__(256)
void k_final_dot(const float* __restrict__ Xf, const float* __restrict__ V2t,
                 float* __restrict__ out) {
  __shared__ float red[256];
  float s = 0.f;
  for (int idx = threadIdx.x; idx < 22528; idx += 256) {
    int m = idx % 88; int d = idx / 88;
    s += Xf[idx] * V2t[(size_t)m * 256 + d];
  }
  red[threadIdx.x] = s; __syncthreads();
  for (int off = 128; off > 0; off >>= 1) {
    if (threadIdx.x < off) red[threadIdx.x] += red[threadIdx.x + off];
    __syncthreads();
  }
  if (threadIdx.x == 0) out[0] = red[0];
}

// ---------------------------------------------------------------------------
// host-side launch helpers
// ---------------------------------------------------------------------------
static inline void launch_gemm(const float* A, int as0, int as1,
                               const float* B, int bs0, int bs1,
                               float* C, const float* bias, int biasMode,
                               int M, int N, int K, int act, hipStream_t s) {
  dim3 grid((N + 63) / 64, (M + 63) / 64);
  k_gemm_wmma<<<grid, 128, 0, s>>>(A, as0, as1, B, bs0, bs1, C, bias, biasMode, M, N, K, act);
}

static inline void launch_conv(const float* X, const float* Wt, const float* bias, float* Y,
                               int Nimg, int Cin, int H, int W, int OC, int KH, int KW,
                               int stride, int pad, int OH, int OW, int relu, hipStream_t s) {
  int cols = Nimg * OH * OW;
  dim3 grid((cols + 63) / 64, (OC + 63) / 64);
  k_conv2d_wmma<<<grid, 128, 0, s>>>(X, Wt, bias, Y, Nimg, Cin, H, W, OC, KH, KW,
                                     stride, pad, OH, OW, relu);
}

extern "C" void kernel_launch(void* const* d_in, const int* in_sizes, int n_in,
                              void* d_out, int out_size, void* d_ws, size_t ws_size,
                              hipStream_t stream) {
  (void)in_sizes; (void)n_in; (void)out_size; (void)ws_size;
  const float* search = (const float*)d_in[0];
  const float* exem   = (const float*)d_in[1];
  const float* aw1 = (const float*)d_in[2];  const float* ab1 = (const float*)d_in[3];
  const float* aw2 = (const float*)d_in[4];  const float* ab2 = (const float*)d_in[5];
  const float* aw3 = (const float*)d_in[6];  const float* ab3 = (const float*)d_in[7];
  const float* aw4 = (const float*)d_in[8];  const float* ab4 = (const float*)d_in[9];
  const float* aw5 = (const float*)d_in[10]; const float* ab5 = (const float*)d_in[11];
  const float* wcd = (const float*)d_in[12]; const float* bcd = (const float*)d_in[13];
  const float* wt  = (const float*)d_in[14]; const float* bt  = (const float*)d_in[15];
  const float* ws1 = (const float*)d_in[16]; const float* bs1 = (const float*)d_in[17];
  const float* ws2 = (const float*)d_in[18]; const float* bs2 = (const float*)d_in[19];
  const float* wg  = (const float*)d_in[20]; const float* bg  = (const float*)d_in[21];
  const float* wh  = (const float*)d_in[22]; const float* bh  = (const float*)d_in[23];
  const float* wc1 = (const float*)d_in[24]; const float* bc1 = (const float*)d_in[25];
  const float* wc2 = (const float*)d_in[26]; const float* bc2 = (const float*)d_in[27];
  float* out = (float*)d_out;

  // ---- workspace layout (floats) ----
  float* W = (float*)d_ws;
  size_t o = 0;
  const size_t IMG = 3ull * 383 * 287;               // 329,763
  float* IN    = W + o; o += 17 * IMG;               // ping buffer (also packed input)
  float* BUF   = W + o; o += 17ull * 64 * 95 * 71;   // pong buffer (largest conv out)
  float* FEAT  = W + o; o += 17ull * 22528;          // pool5: [17][256][88]
  float* NODES = W + o; o += 1408ull * 256;
  float* PB    = W + o; o += 1408ull * 512;          // GEMM products (reused)
  float* HB    = W + o; o += 1408ull * 512;          // GCN hidden (reused)
  float* SB    = W + o; o += 16ull * 512;            // per-timestep column sums
  float* V1    = W + o; o += 22528;
  float* XH    = W + o; o += 22528;
  float* VX    = W + o; o += 22528;
  float* GB    = W + o; o += 22528;
  float* HHB   = W + o; o += 22528;
  float* A2    = W + o; o += 88ull * 88;
  float* C1B   = W + o; o += 88ull * 384;
  float* V2B   = W + o; o += 22528;
  float* VV    = W + o; o += 256;

  // ---- pack search + exemplars into one 17-image batch ----
  hipMemcpyAsync(IN, search, IMG * sizeof(float), hipMemcpyDeviceToDevice, stream);
  hipMemcpyAsync(IN + IMG, exem, 16 * IMG * sizeof(float), hipMemcpyDeviceToDevice, stream);

  // ---- AlexNet features (implicit-GEMM WMMA convs, fused ReLU) ----
  launch_conv(IN,  aw1, ab1, BUF, 17,   3, 383, 287,  64, 11, 11, 4, 2, 95, 71, 1, stream);
  { int tot = 17 * 64 * 47 * 35;
    k_maxpool3x3s2<<<(tot + 255) / 256, 256, 0, stream>>>(BUF, IN, 17 * 64, 95, 71, 47, 35); }
  launch_conv(IN,  aw2, ab2, BUF, 17,  64,  47,  35, 192,  5,  5, 1, 2, 47, 35, 1, stream);
  { int tot = 17 * 192 * 23 * 17;
    k_maxpool3x3s2<<<(tot + 255) / 256, 256, 0, stream>>>(BUF, IN, 17 * 192, 47, 35, 23, 17); }
  launch_conv(IN,  aw3, ab3, BUF, 17, 192,  23,  17, 384,  3,  3, 1, 1, 23, 17, 1, stream);
  launch_conv(BUF, aw4, ab4, IN,  17, 384,  23,  17, 256,  3,  3, 1, 1, 23, 17, 1, stream);
  launch_conv(IN,  aw5, ab5, BUF, 17, 256,  23,  17, 256,  3,  3, 1, 1, 23, 17, 1, stream);
  { int tot = 17 * 256 * 11 * 8;
    k_maxpool3x3s2<<<(tot + 255) / 256, 256, 0, stream>>>(BUF, FEAT, 17 * 256, 23, 17, 11, 8); }

  const float* Xf = FEAT;                  // image 0 = search features [256,88]

  // ---- st-GCN over TM=1408 nodes (A1 applied analytically) ----
  k_build_nodes<<<(1408 * 256 + 255) / 256, 256, 0, stream>>>(FEAT, NODES);
  launch_gemm(NODES, 256, 1, ws1, 512, 1, PB, nullptr, 0, 1408, 512, 256, 0, stream); // P1
  k_colsum_t<<<(16 * 512 + 255) / 256, 256, 0, stream>>>(PB, SB, 512);
  k_a1_apply<<<(1408 * 512 + 255) / 256, 256, 0, stream>>>(PB, SB, bs1, HB, 512);     // h1
  launch_gemm(HB, 512, 1, ws2, 256, 1, PB, nullptr, 0, 1408, 256, 512, 0, stream);    // P2
  k_colsum_t<<<(16 * 256 + 255) / 256, 256, 0, stream>>>(PB, SB, 256);
  k_a1_apply<<<(1408 * 256 + 255) / 256, 256, 0, stream>>>(PB, SB, bs2, HB, 256);     // h2
  k_temporal_max<<<(22528 + 255) / 256, 256, 0, stream>>>(HB, V1);

  // ---- conv_deconv branch ----
  k_conv1d_vmax<<<256, 128, 0, stream>>>(Xf, wcd, bcd, VV);
  k_deconv<<<(22528 + 255) / 256, 256, 0, stream>>>(VV, wt, bt, XH);
  k_add<<<(22528 + 255) / 256, 256, 0, stream>>>(V1, XH, VX, 22528);

  // ---- dynamic cross-track graph ----
  launch_gemm(wg, 256, 1, VX, 88, 1, GB,  bg, 1, 256, 88, 256, 0, stream);    // G
  launch_gemm(wh, 256, 1, VX, 88, 1, HHB, bh, 1, 256, 88, 256, 0, stream);    // Hh
  launch_gemm(HHB, 1, 88, GB, 88, 1, A2, nullptr, 0, 88, 88, 256, 0, stream); // Hh^T @ G
  k_softmax88<<<88, 128, 0, stream>>>(A2);

  // ---- ct-GCN ----
  launch_gemm(V1, 1, 88, wc1, 384, 1, PB, nullptr, 0, 88, 384, 256, 0, stream); // xv@wc1
  launch_gemm(A2, 88, 1, PB, 384, 1, C1B, bc1, 2, 88, 384, 88, 1, stream);      // c1
  launch_gemm(C1B, 384, 1, wc2, 256, 1, PB, nullptr, 0, 88, 256, 384, 0, stream); // c1@wc2
  launch_gemm(A2, 88, 1, PB, 256, 1, V2B, bc2, 2, 88, 256, 88, 1, stream);      // V2^T

  // ---- final full-overlap correlation -> scalar ----
  k_final_dot<<<1, 256, 0, stream>>>(Xf, V2B, out);
}
